// Network_89077621719653
// MI455X (gfx1250) — matrix operand order repaired and checked by reading, not compiled, not run on hardware
//
#include <hip/hip_runtime.h>
#include <hip/hip_bf16.h>

typedef __attribute__((ext_vector_type(16))) _Float16 v16h;
typedef __attribute__((ext_vector_type(8)))  _Float16 v8h;
typedef __attribute__((ext_vector_type(4)))  _Float16 v4h;
typedef __attribute__((ext_vector_type(8)))  float    v8f;
typedef __attribute__((ext_vector_type(4)))  float    f32x4;

#define T_STEPS 1000
#define FB   40
#define L1C  32
#define L2C  20
#define L3C  16
#define NCLS 10
// Fused staging row: [ x(0..39) | h1(40..71) | h2(72..91) | 1.0 @92 | 0 (93..103) ]
// Layer-1 A-window: cols 0..95  (3 chunks; h2 cols meet zero B-rows)
// Layer-2 A-window: cols 40..103 (2 chunks; k_local = col-40, bias @ k=52)
// fc3    A-window: cols 72..103 (1 chunk;  k_local = col-72, bias @ k=20)
#define UW   104  // row stride in halves (208B: 16B-aligned, 2-way LDS banking)
#define BIASCOL 92

static __device__ __forceinline__ v8f wmma16(v16h a, v16h b, v8f c) {
  return __builtin_amdgcn_wmma_f32_16x16x32_f16(false, a, false, b, (short)0, c,
                                                false, false);
}

// A-fragment (16-bit, 16x32): lanes 0-15 take K = klo..klo+7 and klo+16..klo+23;
// lanes 16-31 the same with klo += 8 (caller folds +8 into klo). Two b128 DS loads.
static __device__ __forceinline__ v16h load_a(const _Float16* row, int klo) {
  union { v16h v; v8h p[2]; } u;
  u.p[0] = *(const v8h*)(row + klo);
  u.p[1] = *(const v8h*)(row + klo + 16);
  return u.v;
}

// B-fragment (16-bit, 32x16): lane L holds column n=L%16; lanes 0-15 cover
// K=c*32+0..7,16..23, lanes 16-31 cover K=c*32+8..15,24..31.
// Rows: k<Kw -> weight row k; k==biasK -> bias; else 0.
static __device__ __forceinline__ v16h make_b(const float* __restrict__ w,
                                              const float* __restrict__ bias,
                                              int Kw, int biasK, int Nreal,
                                              int c, int tn, int lane) {
  const int n = tn * 16 + (lane & 15);
  const int kbase = c * 32 + ((lane >> 4) << 3);
  v16h r;
#pragma unroll
  for (int i = 0; i < 16; ++i) {
    const int k = kbase + i + (i >= 8 ? 8 : 0);
    float v = 0.0f;
    if (n < Nreal) {
      if (k < Kw)          v = w[k * Nreal + n];
      else if (k == biasK) v = bias[n];
    }
    r[i] = (_Float16)v;
  }
  return r;
}

// sigmoid via native exp2: one v_mul + v_exp + v_add + v_rcp per element.
static __device__ __forceinline__ float fsigmoid(float x) {
  return __builtin_amdgcn_rcpf(
      1.0f + __builtin_amdgcn_exp2f(-1.442695041f * x));
}
// tanh(x) = 2*sigmoid(2x) - 1 ; sign & 2x folded into the exp2 constant.
// Exact at +/-inf via rcp(inf)=0; no fabs/copysign.
static __device__ __forceinline__ float ftanh(float x) {
  return fmaf(2.0f,
              __builtin_amdgcn_rcpf(
                  1.0f + __builtin_amdgcn_exp2f(-2.885390082f * x)),
              -1.0f);
}

__launch_bounds__(32, 1)
__global__ void gru2_wmma_kernel(const float* __restrict__ x,
                                 const float* __restrict__ w_zx1, const float* __restrict__ b_zx1,
                                 const float* __restrict__ w_hx1, const float* __restrict__ b_hx1,
                                 const float* __restrict__ w_zx2, const float* __restrict__ b_zx2,
                                 const float* __restrict__ w_hx2, const float* __restrict__ b_hx2,
                                 const float* __restrict__ w3, const float* __restrict__ b3,
                                 const float* __restrict__ w4, const float* __restrict__ b4,
                                 float* __restrict__ out) {
  __shared__ __align__(16) _Float16 u[16 * UW];    // fused staging buffer
  __shared__ __align__(16) _Float16 sa[16 * 32];   // fc3 acts [0..15] | 1.0 @16 | 0
  __shared__ __align__(16) float    sred[16 * 16]; // log-softmax staging

  const int lane = threadIdx.x;     // wave32: one wave per block
  const int nl   = lane & 15;
  const int hi   = lane >> 4;
  const int b0   = blockIdx.x * 16;

  // ---- init LDS: zeros, constant-1.0 bias columns. h1_0 = h2_0 = 0 lands here.
  for (int i = lane; i < 16 * UW; i += 32) u[i] = (_Float16)0.0f;
  for (int i = lane; i < 16 * 32; i += 32) sa[i] = (_Float16)0.0f;
  if (lane < 16) {
    u[lane * UW + BIASCOL] = (_Float16)1.0f;
    sa[lane * 32 + L3C]    = (_Float16)1.0f;
  }

  // ---- resident weight fragments ----
  // Layer 1: rows 0..71 = [x|h1] weights, rows 72..91 = 0 (h2 cols), row 92 = bias.
  // Layer 2: k_local rows 0..51 = [h1|h2] weights, row 52 = bias (col 92).
  v16h wz1[3][2], wh1[3][2], wz2[2][2], wh2[2][2];
#pragma unroll
  for (int c = 0; c < 3; ++c)
#pragma unroll
    for (int tn = 0; tn < 2; ++tn) {
      wz1[c][tn] = make_b(w_zx1, b_zx1, FB + L1C, BIASCOL, L1C, c, tn, lane);
      wh1[c][tn] = make_b(w_hx1, b_hx1, FB + L1C, BIASCOL, L1C, c, tn, lane);
    }
#pragma unroll
  for (int c = 0; c < 2; ++c)
#pragma unroll
    for (int tn = 0; tn < 2; ++tn) {
      wz2[c][tn] = make_b(w_zx2, b_zx2, L1C + L2C, L1C + L2C, L2C, c, tn, lane);
      wh2[c][tn] = make_b(w_hx2, b_hx2, L1C + L2C, L1C + L2C, L2C, c, tn, lane);
    }

  // ---- recurrent state in accumulator (C/D) layout ----
  v8f h1t[2] = {};
  v8f h2t[2] = {};

  const int kb = hi * 8;
  const _Float16* arow = u + nl * UW;
  const float* xrow = x + ((size_t)(b0 + nl) * T_STEPS) * FB + hi * 20;

  for (int t = 0; t < T_STEPS; ++t) {
    // -- stage x_t only (h1/h2 were staged at the end of the previous step) --
    const float* xp = xrow + (size_t)t * FB;
    __builtin_prefetch(xp + FB, 0, 1);               // next timestep
    _Float16* up = u + nl * UW + hi * 20;
#pragma unroll
    for (int j = 0; j < 20; j += 4) {
      f32x4 v = *(const f32x4*)(xp + j);
      v4h h = {(_Float16)v.x, (_Float16)v.y, (_Float16)v.z, (_Float16)v.w};
      *(v4h*)(up + j) = h;
    }
    asm volatile("s_wait_dscnt 0" ::: "memory");

    // -- layer 1: K window cols 0..95, bias via col 92, C starts inline-0 --
    const v16h a0 = load_a(arow, 0 + kb);
    const v16h a1 = load_a(arow, 32 + kb);
    const v16h a2 = load_a(arow, 64 + kb);
    v8f zc[2] = {};
    v8f gc[2] = {};
#pragma unroll
    for (int tn = 0; tn < 2; ++tn) {
      zc[tn] = wmma16(a0, wz1[0][tn], zc[tn]);
      zc[tn] = wmma16(a1, wz1[1][tn], zc[tn]);
      zc[tn] = wmma16(a2, wz1[2][tn], zc[tn]);
      gc[tn] = wmma16(a0, wh1[0][tn], gc[tn]);
      gc[tn] = wmma16(a1, wh1[1][tn], gc[tn]);
      gc[tn] = wmma16(a2, wh1[2][tn], gc[tn]);
    }
#pragma unroll
    for (int tile = 0; tile < 2; ++tile)
#pragma unroll
      for (int r = 0; r < 8; ++r) {
        const float z = fsigmoid(zc[tile][r]);
        const float g = ftanh(gc[tile][r]);
        h1t[tile][r] = fmaf(z, g - h1t[tile][r], h1t[tile][r]);
      }

    // -- stage h1_t (serves layer-2 now AND layer-1 of t+1) and h2_{t-1} --
#pragma unroll
    for (int tile = 0; tile < 2; ++tile)
#pragma unroll
      for (int r = 0; r < 8; ++r) {
        const int m = r + 8 * hi;
        u[m * UW + FB + tile * 16 + nl] = (_Float16)h1t[tile][r];
      }
#pragma unroll
    for (int r = 0; r < 8; ++r) {
      const int m = r + 8 * hi;
      u[m * UW + 72 + nl] = (_Float16)h2t[0][r];
      if (nl < 4) u[m * UW + 88 + nl] = (_Float16)h2t[1][r];
    }
    asm volatile("s_wait_dscnt 0" ::: "memory");

    // -- layer 2: K window cols 40..103 --
    const v16h c0 = load_a(arow, 40 + kb);
    const v16h c1 = load_a(arow, 72 + kb);
    v8f z2[2] = {};
    v8f g2[2] = {};
#pragma unroll
    for (int tn = 0; tn < 2; ++tn) {
      z2[tn] = wmma16(c0, wz2[0][tn], z2[tn]);
      z2[tn] = wmma16(c1, wz2[1][tn], z2[tn]);
      g2[tn] = wmma16(c0, wh2[0][tn], g2[tn]);
      g2[tn] = wmma16(c1, wh2[1][tn], g2[tn]);
    }
#pragma unroll
    for (int tile = 0; tile < 2; ++tile)
#pragma unroll
      for (int r = 0; r < 8; ++r) {
        const float z = fsigmoid(z2[tile][r]);
        const float g = ftanh(g2[tile][r]);
        h2t[tile][r] = fmaf(z, g - h2t[tile][r], h2t[tile][r]);
      }
  }

  // ---- epilogue: relu(h2@W3+b3) @ W4 + b4, log_softmax ----
  const v16h w3f = make_b(w3, b3, L2C, L2C, L3C, 0, 0, lane);   // bias @ k=20 <-> col 92
  const v16h w4f = make_b(w4, b4, L3C, L3C, NCLS, 0, 0, lane);  // bias @ k=16 <-> sa col 16

#pragma unroll
  for (int r = 0; r < 8; ++r) {
    const int m = r + 8 * hi;
    u[m * UW + 72 + nl] = (_Float16)h2t[0][r];
    if (nl < 4) u[m * UW + 88 + nl] = (_Float16)h2t[1][r];
  }
  asm volatile("s_wait_dscnt 0" ::: "memory");

  v8f fc3 = {};
  fc3 = wmma16(load_a(arow, 72 + kb), w3f, fc3);    // K window = cols 72..103
#pragma unroll
  for (int r = 0; r < 8; ++r) {
    const int m = r + 8 * hi;
    sa[m * 32 + nl] = (_Float16)fmaxf(fc3[r], 0.0f);
  }
  asm volatile("s_wait_dscnt 0" ::: "memory");

  v8f fo = {};
  fo = wmma16(load_a(sa + nl * 32, kb), w4f, fo);
#pragma unroll
  for (int r = 0; r < 8; ++r) {
    const int m = r + 8 * hi;
    sred[m * 16 + nl] = fo[r];
  }
  asm volatile("s_wait_dscnt 0" ::: "memory");

  if (lane < 16) {
    const float* rowv = sred + lane * 16;
    float mx = rowv[0];
#pragma unroll
    for (int j = 1; j < NCLS; ++j) mx = fmaxf(mx, rowv[j]);
    float s = 0.0f;
#pragma unroll
    for (int j = 0; j < NCLS; ++j) s += __expf(rowv[j] - mx);
    const float lse = mx + __logf(s);
    float* op = out + (size_t)(b0 + lane) * NCLS;
#pragma unroll
    for (int j = 0; j < NCLS; ++j) op[j] = rowv[j] - lse;
  }
}

extern "C" void kernel_launch(void* const* d_in, const int* in_sizes, int n_in,
                              void* d_out, int out_size, void* d_ws, size_t ws_size,
                              hipStream_t stream) {
  const float* x     = (const float*)d_in[0];
  const float* w_zx1 = (const float*)d_in[1];
  const float* b_zx1 = (const float*)d_in[2];
  const float* w_hx1 = (const float*)d_in[3];
  const float* b_hx1 = (const float*)d_in[4];
  const float* w_zx2 = (const float*)d_in[5];
  const float* b_zx2 = (const float*)d_in[6];
  const float* w_hx2 = (const float*)d_in[7];
  const float* b_hx2 = (const float*)d_in[8];
  const float* w3    = (const float*)d_in[9];
  const float* b3    = (const float*)d_in[10];
  const float* w4    = (const float*)d_in[11];
  const float* b4    = (const float*)d_in[12];

  const int Bsz = in_sizes[0] / (T_STEPS * FB);        // 512
  dim3 grid(Bsz / 16), block(32);
  hipLaunchKernelGGL(gru2_wmma_kernel, grid, block, 0, stream,
                     x, w_zx1, b_zx1, w_hx1, b_hx1,
                     w_zx2, b_zx2, w_hx2, b_hx2,
                     w3, b3, w4, b4, (float*)d_out);
}
